// LSTM_RNN_30477087933051
// MI455X (gfx1250) — compile-verified, matrix operand order checked by reading, not run on hardware
//
#include <hip/hip_runtime.h>
#include <hip/hip_bf16.h>
#include <math.h>

// ---------------------------------------------------------------------------
// Problem constants (fixed by the reference)
// ---------------------------------------------------------------------------
#define NTOK 1024
#define HDIM 100
#define GDIM 400      // 4*H, also the N dim of both input GEMMs
#define KP   128      // padded K for WMMA (125 -> 128, 100 -> 128)
#define LDIM 40

typedef __attribute__((ext_vector_type(16))) _Float16 v16h;
typedef __attribute__((ext_vector_type(8)))  _Float16 h8;
typedef __attribute__((ext_vector_type(2)))  _Float16 h2;
typedef __attribute__((ext_vector_type(8)))  float    v8f;

// ---------------------------------------------------------------------------
// Pack kernels: gather embeddings / convert weights to f16 with K padded to 128
// ---------------------------------------------------------------------------
__global__ void pack_emb_kernel(const int* __restrict__ widx, const int* __restrict__ tidx,
                                const float* __restrict__ wtab, const float* __restrict__ ttab,
                                _Float16* __restrict__ X)
{
    int t = blockIdx.x, k = threadIdx.x;          // block = 128 threads
    int w = widx[t], tg = tidx[t];
    float v = 0.f;
    if (k < 100)       v = wtab[w * 100 + k];
    else if (k < 125)  v = ttab[tg * 25 + (k - 100)];
    X[t * KP + k] = (_Float16)v;
}

__global__ void pack_w_kernel(const float* __restrict__ W, _Float16* __restrict__ out, int K)
{
    int r = blockIdx.x, k = threadIdx.x;          // block = 128 threads
    float v = (k < K) ? W[r * K + k] : 0.f;
    out[r * KP + k] = (_Float16)v;
}

// ---------------------------------------------------------------------------
// WMMA GEMM: C[M][400] = A[M][128] x (W[400][128])^T + (bias0 + bias1).
// One 16x16 tile per wave32. All 16 fragment loads (full K=128 for both A and
// B) are issued as one clause, then the 4 WMMAs chain through the accumulator
// with no intervening memory waits.
//
// ISA 7.12.2 16-bit A layout (B is the N<->M mirror == same load applied to
// rows of W):
//   lanes 0-15 : row = lane,    halves 0..7 = K kb+0..7,  8..15 = kb+16..23
//   lanes 16-31: row = lane-16, halves 0..7 = K kb+8..15, 8..15 = kb+24..31
// ---------------------------------------------------------------------------
__global__ void gemm_wmma_kernel(const _Float16* __restrict__ A,
                                 const _Float16* __restrict__ B,
                                 const float* __restrict__ bias0,
                                 const float* __restrict__ bias1,
                                 float* __restrict__ C)
{
    const int ntiles = GDIM >> 4;                  // 25
    int tn = blockIdx.x % ntiles;
    int tm = blockIdx.x / ntiles;
    int lane = threadIdx.x;                        // blockDim.x == 32
    int m0 = tm << 4, n0 = tn << 4;

    // 8-half chunk i of row r sits at K = (lane>>4)*8 + 16*i.
    const _Float16* qa = A + (m0 + (lane & 15)) * KP + ((lane >> 4) << 3);
    const _Float16* qb = B + (n0 + (lane & 15)) * KP + ((lane >> 4) << 3);

    h8 a[8], b[8];
#pragma unroll
    for (int i = 0; i < 8; ++i) a[i] = *(const h8*)(qa + i * 16);
#pragma unroll
    for (int i = 0; i < 8; ++i) b[i] = *(const h8*)(qb + i * 16);

    v8f acc = {};
#pragma unroll
    for (int s = 0; s < 4; ++s) {                  // kb = 32*s uses chunks 2s, 2s+1
        v16h av, bv;
#pragma unroll
        for (int i = 0; i < 8; ++i) {
            av[i] = a[2 * s][i]; av[i + 8] = a[2 * s + 1][i];
            bv[i] = b[2 * s][i]; bv[i + 8] = b[2 * s + 1][i];
        }
        acc = __builtin_amdgcn_wmma_f32_16x16x32_f16(false, av, false, bv,
                                                     (short)0, acc, false, false);
    }

    // C layout: VGPR r -> M = m0 + r + 8*(lane>=16); N = n0 + (lane&15)
    int col   = n0 + (lane & 15);
    float bs  = bias0[col] + bias1[col];
    int rbase = m0 + ((lane >> 4) << 3);
    float* p  = C + rbase * GDIM + col;            // 8 stores, immediate offsets r*1600B
#pragma unroll
    for (int r = 0; r < 8; ++r)
        p[r * GDIM] = acc[r] + bs;
}

// ---------------------------------------------------------------------------
// Sequential LSTM layer. One workgroup; W_hh staged into LDS as k-major
// packed-f16 pairs (80 KB of the WGP's 320 KB). Pre-gates Xg already contain
// x@W_ih^T + b_ih + b_hh. Gate order i,f,g,o.
// ---------------------------------------------------------------------------
#define LSTM_LDS_BYTES (GDIM * 50 * 4 /*h2 pairs*/ + 128 * 4 /*h*/ + GDIM * 4 /*gates*/)

__global__ void lstm_seq_kernel(const float* __restrict__ Xg,    // [n][400]
                                const float* __restrict__ Whh,   // [400][100] f32
                                float* __restrict__ Hout,        // [n][100] f32 or null
                                _Float16* __restrict__ Hf16,     // [n][128] f16 or null
                                int n)
{
    extern __shared__ char smem[];
    h2*    WT2  = (h2*)smem;                          // [50][400], k-major
    float* hbuf = (float*)(smem + GDIM * 50 * 4);     // [128]
    float* gbuf = (float*)(smem + GDIM * 50 * 4 + 128 * 4); // [400]
    int tid = threadIdx.x;

    for (int idx = tid; idx < 50 * GDIM; idx += blockDim.x) {
        int k2 = idx / GDIM, j = idx - k2 * GDIM;
        h2 w;
        w[0] = (_Float16)Whh[j * HDIM + 2 * k2];
        w[1] = (_Float16)Whh[j * HDIM + 2 * k2 + 1];
        WT2[idx] = w;
    }
    if (tid < 128) hbuf[tid] = 0.f;
    float c = 0.f;
    __syncthreads();

    for (int t = 0; t < n; ++t) {
        if (tid < GDIM) {
            if ((tid & 63) == 0 && t + 1 < n)
                __builtin_prefetch(&Xg[(t + 1) * GDIM + tid], 0, 0); // global_prefetch_b8
            float a0 = Xg[t * GDIM + tid], a1 = 0.f, a2 = 0.f, a3 = 0.f;
#pragma unroll
            for (int k2 = 0; k2 < 50; k2 += 2) {
                h2 w0 = WT2[k2 * GDIM + tid];
                h2 w1 = WT2[(k2 + 1) * GDIM + tid];
                float2 hh0 = *(const float2*)&hbuf[2 * k2];
                float2 hh1 = *(const float2*)&hbuf[2 * k2 + 2];
                a0 = fmaf((float)w0[0], hh0.x, a0);
                a1 = fmaf((float)w0[1], hh0.y, a1);
                a2 = fmaf((float)w1[0], hh1.x, a2);
                a3 = fmaf((float)w1[1], hh1.y, a3);
            }
            gbuf[tid] = (a0 + a1) + (a2 + a3);
        }
        __syncthreads();
        if (tid < HDIM) {
            float gi = gbuf[tid], gf = gbuf[HDIM + tid];
            float gg = gbuf[2 * HDIM + tid], go = gbuf[3 * HDIM + tid];
            float si = 1.f / (1.f + __expf(-gi));
            float sf = 1.f / (1.f + __expf(-gf));
            float so = 1.f / (1.f + __expf(-go));
            c = sf * c + si * tanhf(gg);
            float h = so * tanhf(c);
            hbuf[tid] = h;
            if (Hout) Hout[t * HDIM + tid] = h;
            if (Hf16) Hf16[t * KP + tid] = (_Float16)h;
        } else if (tid < 128) {
            if (Hf16) Hf16[t * KP + tid] = (_Float16)0.f;  // K padding for next GEMM
        }
        __syncthreads();
    }
}

// ---------------------------------------------------------------------------
// Rank-1 scorer prep: va = w_sc @ W_h2h[:, :H], vb = w_sc @ W_h2h[:, H:],
// c = b_h2h . w_sc + b_sc
// ---------------------------------------------------------------------------
__global__ void score_prep_kernel(const float* __restrict__ Wh2h,  // [100][200]
                                  const float* __restrict__ bh2h,
                                  const float* __restrict__ wsc,   // [100]
                                  const float* __restrict__ bsc,
                                  float* __restrict__ va, float* __restrict__ vb,
                                  float* __restrict__ scal)
{
    int k = threadIdx.x;                           // block = 128
    if (k < HDIM) {
        float a = 0.f, b = 0.f;
        for (int m = 0; m < HDIM; ++m) {
            float w = wsc[m];
            a = fmaf(w, Wh2h[m * 200 + k], a);
            b = fmaf(w, Wh2h[m * 200 + HDIM + k], b);
        }
        va[k] = a; vb[k] = b;
    }
    if (k == 0) {
        float cc = bsc[0];
        for (int m = 0; m < HDIM; ++m) cc = fmaf(bh2h[m], wsc[m], cc);
        scal[0] = cc;
    }
}

// sA/sB matvecs + exponentials + block reduction of S = sum_i exp(sA[i])
__global__ void sasb_kernel(const float* __restrict__ H1, const float* __restrict__ va,
                            const float* __restrict__ vb, const float* __restrict__ scal,
                            float* __restrict__ eA, float* __restrict__ eB,
                            float* __restrict__ Sout)
{
    __shared__ float red[NTOK];
    int i = threadIdx.x;                           // block = 1024
    const float* h = H1 + i * HDIM;
    float sa = 0.f, sb = 0.f;
    for (int k = 0; k < HDIM; ++k) {
        float hv = h[k];
        sa = fmaf(hv, va[k], sa);
        sb = fmaf(hv, vb[k], sb);
    }
    float ea = __expf(sa);
    float eb = __expf(sb + scal[0]);
    eA[i] = ea; eB[i] = eb;
    red[i] = ea; __syncthreads();
    for (int s = NTOK / 2; s > 0; s >>= 1) {
        if (i < s) red[i] += red[i + s];
        __syncthreads();
    }
    if (i == 0) Sout[0] = red[0];
}

// scores[i,j] = eA[i]*eB[j] / (eB[j]*S + eps)  — pure streaming write (4 MB)
__global__ void scores_kernel(const float* __restrict__ eA, const float* __restrict__ eB,
                              const float* __restrict__ Sp, float* __restrict__ out)
{
    int idx = blockIdx.x * blockDim.x + threadIdx.x;
    int i = idx >> 10, j = idx & (NTOK - 1);
    float eb = eB[j];
    out[idx] = eA[i] * eb / fmaf(eb, Sp[0], 1e-20f);
}

// labels: per-row exp-normalize of h[1:] @ W_lab^T + b_lab
__global__ void labels_kernel(const float* __restrict__ H1, const float* __restrict__ Wlab,
                              const float* __restrict__ blab, float* __restrict__ out)
{
    __shared__ float es[LDIM];
    __shared__ float ssum;
    int row = blockIdx.x + 1;
    int l = threadIdx.x;                           // block = 64
    float e = 0.f;
    if (l < LDIM) {
        float acc = blab[l];
        const float* h = H1 + row * HDIM;
        const float* w = Wlab + l * HDIM;
        for (int k = 0; k < HDIM; ++k) acc = fmaf(h[k], w[k], acc);
        e = __expf(acc);
        es[l] = e;
    }
    __syncthreads();
    if (l == 0) { float s = 0.f; for (int q = 0; q < LDIM; ++q) s += es[q]; ssum = s; }
    __syncthreads();
    if (l < LDIM) out[blockIdx.x * LDIM + l] = e / (ssum + 1e-20f);
}

// ---------------------------------------------------------------------------
// Host launcher
// ---------------------------------------------------------------------------
extern "C" void kernel_launch(void* const* d_in, const int* in_sizes, int n_in,
                              void* d_out, int out_size, void* d_ws, size_t ws_size,
                              hipStream_t stream)
{
    const int*   word_idx = (const int*)  d_in[0];
    const int*   tag_idx  = (const int*)  d_in[1];
    const float* wtab     = (const float*)d_in[2];
    const float* ttab     = (const float*)d_in[3];
    const float* W_ih0    = (const float*)d_in[4];
    const float* W_hh0    = (const float*)d_in[5];
    const float* b_ih0    = (const float*)d_in[6];
    const float* b_hh0    = (const float*)d_in[7];
    const float* W_ih1    = (const float*)d_in[8];
    const float* W_hh1    = (const float*)d_in[9];
    const float* b_ih1    = (const float*)d_in[10];
    const float* b_hh1    = (const float*)d_in[11];
    const float* W_h2h    = (const float*)d_in[12];
    const float* b_h2h    = (const float*)d_in[13];
    const float* W_sc     = (const float*)d_in[14];
    const float* b_sc     = (const float*)d_in[15];
    const float* W_lab    = (const float*)d_in[16];
    const float* b_lab    = (const float*)d_in[17];
    float* out = (float*)d_out;

    const int n = in_sizes[0];                     // 1024

    // Workspace layout (256B aligned)
    char* ws = (char*)d_ws;
    _Float16* Xf16  = (_Float16*)(ws + 0);                       // [1024][128]  256 KB
    _Float16* Wf0   = (_Float16*)(ws + 262144);                  // [400][128]   100 KB
    _Float16* Wf1   = (_Float16*)(ws + 364544);                  // [400][128]   100 KB
    float*    Xg    = (float*)   (ws + 466944);                  // [1024][400]  1.6 MB (reused)
    _Float16* H0f16 = (_Float16*)(ws + 2105344);                 // [1024][128]  256 KB
    float*    H1    = (float*)   (ws + 2367488);                 // [1024][100]  400 KB
    float*    va    = (float*)   (ws + 2777088);
    float*    vb    = (float*)   (ws + 2777600);
    float*    eA    = (float*)   (ws + 2778112);
    float*    eB    = (float*)   (ws + 2782208);
    float*    scal  = (float*)   (ws + 2786304);                 // [0]=c const, [1]=S

    // Allow 80 KB dynamic LDS for the recurrent kernel (WGP has 320 KB).
    hipFuncSetAttribute((const void*)lstm_seq_kernel,
                        hipFuncAttributeMaxDynamicSharedMemorySize, LSTM_LDS_BYTES);

    const int gemm_blocks = (n / 16) * (GDIM / 16);              // 64 * 25

    // 1) pack inputs to f16 (K padded to 128)
    pack_emb_kernel<<<dim3(n), dim3(128), 0, stream>>>(word_idx, tag_idx, wtab, ttab, Xf16);
    pack_w_kernel<<<dim3(GDIM), dim3(128), 0, stream>>>(W_ih0, Wf0, 125);
    pack_w_kernel<<<dim3(GDIM), dim3(128), 0, stream>>>(W_ih1, Wf1, HDIM);

    // 2) layer-0 input GEMM (WMMA) + bias, then sequential recurrence
    gemm_wmma_kernel<<<dim3(gemm_blocks), dim3(32), 0, stream>>>(Xf16, Wf0, b_ih0, b_hh0, Xg);
    lstm_seq_kernel<<<dim3(1), dim3(512), LSTM_LDS_BYTES, stream>>>(Xg, W_hh0, nullptr, H0f16, n);

    // 3) layer-1 input GEMM (WMMA) + bias, then sequential recurrence
    gemm_wmma_kernel<<<dim3(gemm_blocks), dim3(32), 0, stream>>>(H0f16, Wf1, b_ih1, b_hh1, Xg);
    lstm_seq_kernel<<<dim3(1), dim3(512), LSTM_LDS_BYTES, stream>>>(Xg, W_hh1, H1, nullptr, n);

    // 4) rank-1 collapsed pairwise scorer + per-column softmax
    score_prep_kernel<<<dim3(1), dim3(128), 0, stream>>>(W_h2h, b_h2h, W_sc, b_sc, va, vb, scal);
    sasb_kernel<<<dim3(1), dim3(NTOK), 0, stream>>>(H1, va, vb, scal, eA, eB, scal + 1);
    scores_kernel<<<dim3((n * n) / 256), dim3(256), 0, stream>>>(eA, eB, scal + 1, out);

    // 5) label head
    labels_kernel<<<dim3(n - 1), dim3(64), 0, stream>>>(H1, W_lab, b_lab, out + n * n);
}
